// Block_41609643164064
// MI455X (gfx1250) — compile-verified, matrix operand order checked by reading, not run on hardware
//
#include <hip/hip_runtime.h>
#include <hip/hip_bf16.h>
#include <math.h>

// ---------------- problem constants ----------------
#define M_VIEWS 4
#define C_DIM   8
#define K_DIM   16
#define N_PTS   1024
#define D_DIM   96
#define H_DIM   64
#define W_DIM   64
#define HW      (H_DIM * W_DIM)
#define DFF     192
#define TK      64
#define KN      (K_DIM * N_PTS)            // 16384 per (b,c)
#define PER_M   (C_DIM * K_DIM * N_PTS)    // 131072 points per view
#define GROUPS  (M_VIEWS * C_DIM * K_DIM)  // 512 (m,c,k) groups

typedef __attribute__((ext_vector_type(16))) __bf16 v16bf;
typedef __attribute__((ext_vector_type(8)))  float  v8f;
typedef __attribute__((ext_vector_type(2), aligned(4))) float f32x2;   // 4B-aligned pair
typedef __attribute__((ext_vector_type(2))) unsigned int u32x2;

// ---------------------------------------------------------------------------
// K1: qn = LN(query); q = qn@Wq + bq; qhat = l2norm(q).  128 rows, one block.
// ---------------------------------------------------------------------------
__global__ void k1_query(const float* __restrict__ query, const float* __restrict__ Wq,
                         const float* __restrict__ bq, const float* __restrict__ g1,
                         const float* __restrict__ b1ln, float* __restrict__ qhat) {
  const int r = threadIdx.x;                 // 0..127 == c*16+k
  const float* x = query + r * D_DIM;
  float mu = 0.f;
  for (int e = 0; e < D_DIM; ++e) mu += x[e];
  mu *= (1.f / D_DIM);
  float var = 0.f;
  for (int e = 0; e < D_DIM; ++e) { float d = x[e] - mu; var += d * d; }
  var *= (1.f / D_DIM);
  const float rstd = rsqrtf(var + 1e-5f);
  float qn[D_DIM];
  for (int e = 0; e < D_DIM; ++e) qn[e] = (x[e] - mu) * rstd * g1[e] + b1ln[e];
  float q[D_DIM];
  float ss = 0.f;
  for (int d = 0; d < D_DIM; ++d) {
    float acc = bq[d];
    for (int e = 0; e < D_DIM; ++e) acc += qn[e] * Wq[e * D_DIM + d];
    q[d] = acc; ss += acc * acc;
  }
  const float inv = 1.f / fmaxf(sqrtf(ss), 1e-12f);
  for (int d = 0; d < D_DIM; ++d) qhat[r * D_DIM + d] = q[d] * inv;
}

// ---------------------------------------------------------------------------
// K2: heavy pass. One block per (m, c, k); 8 waves; each wave owns 16-point
// tiles. Bilinear-sample cf -> bf16 LDS tile (packed b64 stores), then
// kf = cf@Wk via v_wmma_f32_16x16x32_bf16 with inline-0 C accumulators,
// bias folded into the epilogue, logits = scale * (qhat . kf/||kf||).
// ---------------------------------------------------------------------------
__global__ void __launch_bounds__(256) k2_logits(
    const float* __restrict__ voxels, const float* __restrict__ feats,
    const float* __restrict__ P, const float* __restrict__ Wk,
    const float* __restrict__ bk, const float* __restrict__ lsp,
    const float* __restrict__ qhat, float* __restrict__ logits) {
  __shared__ __align__(64) __bf16 WkT[D_DIM * D_DIM];       // Wk[m] transposed (col-major)
  __shared__ __align__(64) __bf16 cfT[8][16][D_DIM];        // per-wave 16xD cf tile
  __shared__ float qh[D_DIM];
  __shared__ float bks[D_DIM];

  const int g    = blockIdx.x;            // 0..511
  const int m    = g >> 7;                // view
  const int ck   = g & 127;               // c*16 + k
  const int tid  = threadIdx.x;
  const int lane = tid & 31;
  const int wave = tid >> 5;
  const int half = lane >> 4;             // 0/1
  const int pl   = lane & 15;             // point-in-tile / column-in-tile

  // stage Wk[m]^T and per-group vectors into LDS
  const float* wkm = Wk + m * D_DIM * D_DIM;
  for (int i = tid; i < D_DIM * D_DIM; i += 256) {
    const int e = i / D_DIM, d = i % D_DIM;
    WkT[d * D_DIM + e] = (__bf16)wkm[e * D_DIM + d];
  }
  for (int i = tid; i < D_DIM; i += 256) {
    qh[i]  = qhat[ck * D_DIM + i];
    bks[i] = bk[m * D_DIM + i];
  }
  __syncthreads();

  const float p00 = P[m * 6 + 0], p01 = P[m * 6 + 1];
  const float p10 = P[m * 6 + 2], p11 = P[m * 6 + 3];
  const float p20 = P[m * 6 + 4], p21 = P[m * 6 + 5];
  const float scale = __expf(fminf(lsp[0], 4.6051702f));   // exp(min(ls, log(100)))
  const float* fm = feats + m * (D_DIM * HW);

  for (int it = 0; it < 8; ++it) {
    const int tile  = it * 8 + wave;       // 0..63
    const int nbase = tile * 16;

    // ---- bilinear sample: lanes 0-15 -> channels 0..47, lanes 16-31 -> 48..95
    const int n = nbase + pl;
    const float* vx = voxels + (ck * N_PTS + n) * 3;
    const float v0 = vx[0], v1 = vx[1], v2 = vx[2];
    const float gx = v0 * p00 + v1 * p10 + v2 * p20;
    const float gy = v0 * p01 + v1 * p11 + v2 * p21;
    const float bound = (fabsf(gx) <= 1.f && fabsf(gy) <= 1.f) ? 1.f : 0.f;
    const float fx = fminf(fmaxf((gx + 1.f) * 0.5f * (W_DIM - 1), 0.f), (float)(W_DIM - 1));
    const float fy = fminf(fmaxf((gy + 1.f) * 0.5f * (H_DIM - 1), 0.f), (float)(H_DIM - 1));
    const int x0 = (int)fminf(fmaxf(floorf(fx), 0.f), (float)(W_DIM - 2));
    const int y0 = (int)fminf(fmaxf(floorf(fy), 0.f), (float)(H_DIM - 2));
    const float wx = fx - x0, wy = fy - y0;
    const float w00 = (1.f - wx) * (1.f - wy) * bound, w01 = wx * (1.f - wy) * bound;
    const float w10 = (1.f - wx) * wy * bound,         w11 = wx * wy * bound;
    const int o00 = y0 * W_DIM + x0;

    #pragma unroll
    for (int cc = 0; cc < 48; cc += 4) {
      const int ch = half * 48 + cc;
      const float* fp0 = fm + ch * HW + o00;
      if (cc < 44) __builtin_prefetch(fp0 + 4 * HW, 0, 3);   // next 4-channel group
      float s4[4];
      #pragma unroll
      for (int j = 0; j < 4; ++j) {
        const float* fp = fp0 + j * HW;
        const f32x2 t0 = *reinterpret_cast<const f32x2*>(fp);           // taps (y0,x0),(y0,x1)
        const f32x2 t1 = *reinterpret_cast<const f32x2*>(fp + W_DIM);   // taps (y1,x0),(y1,x1)
        s4[j] = t0.x * w00 + t0.y * w01 + t1.x * w10 + t1.y * w11;
      }
      const unsigned short u0 = __builtin_bit_cast(unsigned short, (__bf16)s4[0]);
      const unsigned short u1 = __builtin_bit_cast(unsigned short, (__bf16)s4[1]);
      const unsigned short u2 = __builtin_bit_cast(unsigned short, (__bf16)s4[2]);
      const unsigned short u3 = __builtin_bit_cast(unsigned short, (__bf16)s4[3]);
      const u32x2 packed = { (unsigned)u0 | ((unsigned)u1 << 16),
                             (unsigned)u2 | ((unsigned)u3 << 16) };
      *reinterpret_cast<u32x2*>(&cfT[wave][pl][ch]) = packed;           // ds_store_b64
    }
    // wave-local LDS producer->consumer sync (each wave owns its own tile)
    asm volatile("s_wait_dscnt 0" ::: "memory");

    // ---- kf = cf @ Wk via bf16 WMMA, zero C (inline-0 operand), f32 accumulate
    v8f acc[6] = {};
    #pragma unroll
    for (int kc = 0; kc < 3; ++kc) {
      const v16bf a = *reinterpret_cast<const v16bf*>(&cfT[wave][pl][kc * 32 + half * 16]);
      #pragma unroll
      for (int ct = 0; ct < 6; ++ct) {
        const v16bf b = *reinterpret_cast<const v16bf*>(
            &WkT[(ct * 16 + pl) * D_DIM + kc * 32 + half * 16]);
        acc[ct] = __builtin_amdgcn_wmma_f32_16x16x32_bf16(
            false, a, false, b, (short)0, acc[ct], false, false);
      }
    }

    // ---- epilogue: add bias, per-row ||kf||^2 and qhat.kf. C-layout:
    // lane holds column pl, rows {half*8 + r}; reduce 16 columns per half-wave.
    float ssum[8], dsum[8];
    #pragma unroll
    for (int r = 0; r < 8; ++r) { ssum[r] = 0.f; dsum[r] = 0.f; }
    #pragma unroll
    for (int ct = 0; ct < 6; ++ct) {
      const float qv = qh[ct * 16 + pl];
      const float bb = bks[ct * 16 + pl];
      #pragma unroll
      for (int r = 0; r < 8; ++r) {
        const float v = acc[ct][r] + bb;
        ssum[r] += v * v;
        dsum[r] += v * qv;
      }
    }
    #pragma unroll
    for (int msk = 1; msk < 16; msk <<= 1) {
      #pragma unroll
      for (int r = 0; r < 8; ++r) {
        ssum[r] += __shfl_xor(ssum[r], msk, 32);
        dsum[r] += __shfl_xor(dsum[r], msk, 32);
      }
    }
    if (pl == 0) {
      #pragma unroll
      for (int r = 0; r < 8; ++r) {
        const int row = half * 8 + r;
        const float lg = scale * dsum[r] / fmaxf(sqrtf(ssum[r]), 1e-12f);
        logits[g * N_PTS + nbase + row] = lg;
      }
    }
  }
}

// ---------------------------------------------------------------------------
// K3: softmax over N=1024 per (m,c,k) row, in place.
// ---------------------------------------------------------------------------
__global__ void k3_softmax(float* __restrict__ logits) {
  __shared__ float red[256];
  float* p = logits + blockIdx.x * N_PTS;
  const int tid = threadIdx.x;
  float v[4];
  float mx = -1e30f;
  for (int i = 0; i < 4; ++i) { v[i] = p[tid + i * 256]; mx = fmaxf(mx, v[i]); }
  red[tid] = mx; __syncthreads();
  for (int s = 128; s > 0; s >>= 1) {
    if (tid < s) red[tid] = fmaxf(red[tid], red[tid + s]);
    __syncthreads();
  }
  mx = red[0]; __syncthreads();
  float sum = 0.f;
  for (int i = 0; i < 4; ++i) { v[i] = __expf(v[i] - mx); sum += v[i]; }
  red[tid] = sum; __syncthreads();
  for (int s = 128; s > 0; s >>= 1) {
    if (tid < s) red[tid] += red[tid + s];
    __syncthreads();
  }
  const float inv = 1.f / red[0];
  for (int i = 0; i < 4; ++i) p[tid + i * 256] = v[i] * inv;
}

// K3b: voxel_scores = sum over views.
__global__ void k3b_vscores(const float* __restrict__ scores, float* __restrict__ vs) {
  const int i = blockIdx.x * 256 + threadIdx.x;            // < 131072
  float s = 0.f;
  for (int m = 0; m < M_VIEWS; ++m) s += scores[m * PER_M + i];
  vs[i] = s;
}

// ---------------------------------------------------------------------------
// K4: top-64 per (b,c) over 16384 scores, descending, ties -> lower index.
// Iterative block argmax with an LDS exclusion bitmask.
// ---------------------------------------------------------------------------
__global__ void k4_topk(const float* __restrict__ vs, int* __restrict__ wsidx,
                        int* __restrict__ outi) {
  __shared__ unsigned int msk[KN / 32];
  __shared__ float rv[256];
  __shared__ int   ri[256];
  const int c = blockIdx.x;
  const float* row = vs + c * KN;
  const int tid = threadIdx.x;
  for (int i = tid; i < KN / 32; i += 256) msk[i] = 0u;
  __syncthreads();
  for (int t = 0; t < TK; ++t) {
    float bvv = -1e30f; int bi = KN;
    for (int i = tid; i < KN; i += 256) {
      if (msk[i >> 5] & (1u << (i & 31))) continue;
      const float v = row[i];
      if (v > bvv || (v == bvv && i < bi)) { bvv = v; bi = i; }
    }
    rv[tid] = bvv; ri[tid] = bi;
    __syncthreads();
    for (int s = 128; s > 0; s >>= 1) {
      if (tid < s) {
        const float ov = rv[tid + s]; const int oi = ri[tid + s];
        if (ov > rv[tid] || (ov == rv[tid] && oi < ri[tid])) { rv[tid] = ov; ri[tid] = oi; }
      }
      __syncthreads();
    }
    if (tid == 0) {
      const int b = ri[0];
      msk[b >> 5] |= (1u << (b & 31));
      wsidx[c * TK + t] = b;
      outi[c * TK + t]              = b / N_PTS;   // space_idx
      outi[C_DIM * TK + c * TK + t] = b % N_PTS;   // voxel_idx
    }
    __syncthreads();
  }
}

// ---------------------------------------------------------------------------
// K5: for each selected point: recompute cf per view, vf = cf@Wv+bv,
// o = sum_m s*vf; z = o@Wo+bo; x = z + query[k']; LN2; GeLU FFN; write x.
// One block (192 threads) per selected row.
// ---------------------------------------------------------------------------
__global__ void __launch_bounds__(192) k5_final(
    const float* __restrict__ voxels, const float* __restrict__ query,
    const float* __restrict__ feats, const float* __restrict__ P,
    const float* __restrict__ Wv, const float* __restrict__ bv,
    const float* __restrict__ Wo, const float* __restrict__ bo,
    const float* __restrict__ g2, const float* __restrict__ b2ln,
    const float* __restrict__ W1, const float* __restrict__ b1f,
    const float* __restrict__ W2, const float* __restrict__ b2f,
    const float* __restrict__ scores, const int* __restrict__ wsidx,
    float* __restrict__ xout) {
  __shared__ float cfm[D_DIM];
  __shared__ float os[D_DIM];
  __shared__ float xs[D_DIM];
  __shared__ float xn[D_DIM];
  __shared__ float hh[DFF];
  __shared__ float stat[2];

  const int s = blockIdx.x;                 // 0..511
  const int c = s / TK, t = s % TK;
  const int idx = wsidx[c * TK + t];
  const int kq = idx / N_PTS, nq = idx % N_PTS;
  const int tid = threadIdx.x;

  const float* vx = voxels + ((c * K_DIM + kq) * N_PTS + nq) * 3;
  const float v0 = vx[0], v1 = vx[1], v2 = vx[2];

  float accO = 0.f;
  for (int m = 0; m < M_VIEWS; ++m) {
    if (tid < D_DIM) {
      const float gx = v0 * P[m * 6 + 0] + v1 * P[m * 6 + 2] + v2 * P[m * 6 + 4];
      const float gy = v0 * P[m * 6 + 1] + v1 * P[m * 6 + 3] + v2 * P[m * 6 + 5];
      const float bound = (fabsf(gx) <= 1.f && fabsf(gy) <= 1.f) ? 1.f : 0.f;
      const float fx = fminf(fmaxf((gx + 1.f) * 0.5f * (W_DIM - 1), 0.f), (float)(W_DIM - 1));
      const float fy = fminf(fmaxf((gy + 1.f) * 0.5f * (H_DIM - 1), 0.f), (float)(H_DIM - 1));
      const int x0 = (int)fminf(fmaxf(floorf(fx), 0.f), (float)(W_DIM - 2));
      const int y0 = (int)fminf(fmaxf(floorf(fy), 0.f), (float)(H_DIM - 2));
      const float wx = fx - x0, wy = fy - y0;
      const float* fp = feats + (m * D_DIM + tid) * HW + y0 * W_DIM + x0;
      const f32x2 t0 = *reinterpret_cast<const f32x2*>(fp);
      const f32x2 t1 = *reinterpret_cast<const f32x2*>(fp + W_DIM);
      const float sv = t0.x * (1.f - wx) * (1.f - wy) + t0.y * wx * (1.f - wy)
                     + t1.x * (1.f - wx) * wy + t1.y * wx * wy;
      cfm[tid] = sv * bound;
    }
    __syncthreads();
    if (tid < D_DIM) {
      const float* wvm = Wv + m * D_DIM * D_DIM;
      float a = bv[m * D_DIM + tid];
      for (int e = 0; e < D_DIM; ++e) a += cfm[e] * wvm[e * D_DIM + tid];
      const float sm = scores[m * PER_M + c * KN + idx];
      accO += sm * a;
    }
    __syncthreads();
  }
  if (tid < D_DIM) os[tid] = accO;
  __syncthreads();
  float xv = 0.f;
  if (tid < D_DIM) {
    float z = bo[tid];
    for (int e = 0; e < D_DIM; ++e) z += os[e] * Wo[e * D_DIM + tid];
    xv = z + query[(c * K_DIM + kq) * D_DIM + tid];
    xs[tid] = xv;
  }
  __syncthreads();
  if (tid == 0) {
    float mu = 0.f;
    for (int e = 0; e < D_DIM; ++e) mu += xs[e];
    mu *= (1.f / D_DIM);
    float var = 0.f;
    for (int e = 0; e < D_DIM; ++e) { const float d = xs[e] - mu; var += d * d; }
    stat[0] = mu; stat[1] = rsqrtf(var / D_DIM + 1e-5f);
  }
  __syncthreads();
  if (tid < D_DIM) xn[tid] = (xs[tid] - stat[0]) * stat[1] * g2[tid] + b2ln[tid];
  __syncthreads();
  {
    float h = b1f[tid];                               // all 192 threads: one hidden unit each
    for (int e = 0; e < D_DIM; ++e) h += xn[e] * W1[e * DFF + tid];
    hh[tid] = 0.5f * h * (1.f + erff(h * 0.70710678f));   // exact GELU
  }
  __syncthreads();
  if (tid < D_DIM) {
    float a = b2f[tid];
    for (int f = 0; f < DFF; ++f) a += hh[f] * W2[f * D_DIM + tid];
    xout[(c * TK + t) * D_DIM + tid] = xv + a;
  }
}

// ---------------------------------------------------------------------------
extern "C" void kernel_launch(void* const* d_in, const int* in_sizes, int n_in,
                              void* d_out, int out_size, void* d_ws, size_t ws_size,
                              hipStream_t stream) {
  const float* voxels = (const float*)d_in[0];
  const float* query  = (const float*)d_in[1];
  const float* feats  = (const float*)d_in[2];
  const float* P      = (const float*)d_in[3];
  const float* Wq     = (const float*)d_in[4];
  const float* bq     = (const float*)d_in[5];
  const float* Wk     = (const float*)d_in[6];
  const float* bk     = (const float*)d_in[7];
  const float* Wv     = (const float*)d_in[8];
  const float* bv     = (const float*)d_in[9];
  const float* lsp    = (const float*)d_in[10];
  const float* Wo     = (const float*)d_in[11];
  const float* bo     = (const float*)d_in[12];
  const float* g1     = (const float*)d_in[13];
  const float* b1ln   = (const float*)d_in[14];
  const float* g2     = (const float*)d_in[15];
  const float* b2ln   = (const float*)d_in[16];
  const float* W1     = (const float*)d_in[17];
  const float* b1f    = (const float*)d_in[18];
  const float* W2     = (const float*)d_in[19];
  const float* b2f    = (const float*)d_in[20];

  // workspace layout (floats): qhat | logits/scores | voxel_scores | idx
  float* ws     = (float*)d_ws;
  float* qhat   = ws;                                     // 128*96      = 12288
  float* logits = ws + 12288;                             // 4*131072    = 524288
  float* vsc    = ws + 12288 + 524288;                    // 131072
  int*   wsidx  = (int*)(ws + 12288 + 524288 + 131072);   // 512 ints

  float* xout = (float*)d_out;                            // 8*64*96 = 49152 floats
  int*   outi = ((int*)d_out) + C_DIM * TK * D_DIM;       // then 512+512 int32

  k1_query  <<<1, 128, 0, stream>>>(query, Wq, bq, g1, b1ln, qhat);
  k2_logits <<<GROUPS, 256, 0, stream>>>(voxels, feats, P, Wk, bk, lsp, qhat, logits);
  k3_softmax<<<GROUPS, 256, 0, stream>>>(logits);
  k3b_vscores<<<PER_M / 256, 256, 0, stream>>>(logits, vsc);
  k4_topk   <<<C_DIM, 256, 0, stream>>>(vsc, wsidx, outi);
  k5_final  <<<C_DIM * TK, 192, 0, stream>>>(voxels, query, feats, P, Wv, bv, Wo, bo,
                                             g2, b2ln, W1, b1f, W2, b2f,
                                             logits, wsidx, xout);
}
